// LatDeformSplat2xUp_67224828117115
// MI455X (gfx1250) — matrix operand order repaired and checked by reading, not compiled, not run on hardware
//
#include <hip/hip_runtime.h>
#include <math.h>

// ---------------- problem constants ----------------
#define H_FULL 512
#define W_FULL 1024
#define HP 256
#define WP 512
#define HW (HP*WP)            // 131072  (low-res spatial)
#define HW2 (H_FULL*W_FULL)   // 524288  (high-res spatial)
#define NB 2
#define CIN 128
#define COUT 64
#define NTAP 9
#define GROUPS 32
#define MPTS (HW*NTAP)        // 1179648 splat points
#define MROWS (NTAP*COUT)     // 576 rows of GEMM1

typedef __attribute__((ext_vector_type(16))) _Float16 v16h;
typedef __attribute__((ext_vector_type(8)))  _Float16 v8h;
typedef __attribute__((ext_vector_type(8)))  float    v8f;

// Build a v16h WMMA fragment from two contiguous 16-byte runs.
__device__ __forceinline__ v16h ld_frag2(const _Float16* p0, const _Float16* p1) {
  v8h a = *(const v8h*)p0;
  v8h b = *(const v8h*)p1;
  return __builtin_shufflevector(a, b, 0,1,2,3,4,5,6,7,8,9,10,11,12,13,14,15);
}

__device__ __forceinline__ float gelu_exact(float x) {
  return 0.5f * x * (1.0f + erff(x * 0.70710678118654752440f));
}

// ---------------- kernel 1: GN1 stats (per b,group of 4 channels) ----------------
__global__ void gn1_stats(const float* __restrict__ x, float* __restrict__ stats) {
  __shared__ float s1[256], s2[256];
  int blk = blockIdx.x;             // 0..63
  int b = blk >> 5, g = blk & 31;
  const float* base = x + ((size_t)(b * CIN + g * 4)) * HW;   // 4*HW contiguous
  float a1 = 0.f, a2 = 0.f;
  for (int i = threadIdx.x; i < 4 * HW; i += 256) { float v = base[i]; a1 += v; a2 += v * v; }
  s1[threadIdx.x] = a1; s2[threadIdx.x] = a2; __syncthreads();
  for (int off = 128; off > 0; off >>= 1) {
    if (threadIdx.x < off) { s1[threadIdx.x] += s1[threadIdx.x + off]; s2[threadIdx.x] += s2[threadIdx.x + off]; }
    __syncthreads();
  }
  if (threadIdx.x == 0) {
    float inv = 1.f / (4.f * HW);
    float mu = s1[0] * inv;
    float var = s2[0] * inv - mu * mu;
    stats[blk * 2]     = mu;
    stats[blk * 2 + 1] = rsqrtf(var + 1e-5f);
  }
}

// ---------------- kernel 2: apply GN1, transpose to K-contiguous f16 ----------------
// Output: ht[(b*HW + s)*128 + c]
__global__ void gn1_apply_t(const float* __restrict__ x, const float* __restrict__ stats,
                            const float* __restrict__ gamma, const float* __restrict__ beta,
                            _Float16* __restrict__ ht) {
  __shared__ _Float16 tile[CIN * 32];
  int b  = blockIdx.x >> 12;                 // HW/32 = 4096 tiles per batch
  int s0 = (blockIdx.x & 4095) * 32;
  for (int i = threadIdx.x; i < CIN * 32; i += 256) {
    int c = i >> 5, sl = i & 31;
    float mu = stats[(b * GROUPS + (c >> 2)) * 2];
    float rs = stats[(b * GROUPS + (c >> 2)) * 2 + 1];
    float v  = x[((size_t)(b * CIN + c)) * HW + s0 + sl];
    tile[i] = (_Float16)((v - mu) * rs * gamma[c] + beta[c]);
  }
  __syncthreads();
  for (int i = threadIdx.x; i < CIN * 32; i += 256) {
    int sl = i >> 7, c = i & 127;
    ht[((size_t)(b * HW + s0 + sl)) * CIN + c] = tile[c * 32 + sl];
  }
}

// ---------------- kernel 3: convert both weight matrices to f16 ----------------
__global__ void convert_w(const float* __restrict__ wc, const float* __restrict__ wp,
                          _Float16* __restrict__ wc16, _Float16* __restrict__ wp16) {
  int i = blockIdx.x * 256 + threadIdx.x;
  if (i < MROWS * CIN) wc16[i] = (_Float16)wc[i];
  if (i < COUT * COUT) wp16[i] = (_Float16)wp[i];
}

// ---------------- kernel 4: zero fill ----------------
__global__ void zero_f32(float* __restrict__ p, long n) {
  long i = (long)blockIdx.x * 256 + threadIdx.x;
  long stride = (long)gridDim.x * 256;
  for (; i < n; i += stride) p[i] = 0.f;
}

// ---------------- kernel 5: splat prep (indices, weights, denominator) ----------------
__global__ void splat_prep(const float* __restrict__ grid, int4* __restrict__ idx4,
                           float4* __restrict__ w4, float* __restrict__ den) {
  int m = blockIdx.x * 256 + threadIdx.x;
  if (m >= MPTS) return;
  float gx = grid[2 * m], gy = grid[2 * m + 1];
  float px = (gx + 1.f) * 0.5f * (float)(W_FULL - 1);
  float py = (gy + 1.f) * 0.5f * (float)(H_FULL - 1);
  int x0 = (int)floorf(px), y0 = (int)floorf(py);
  float fx = px - (float)x0, fy = py - (float)y0;
  float wt[4] = { (1.f - fx) * (1.f - fy), (1.f - fx) * fy, fx * (1.f - fy), fx * fy };
  int x0w = x0 & (W_FULL - 1);          // W is a power of two -> correct positive mod
  int x1w = (x0 + 1) & (W_FULL - 1);
  int ys[4] = { y0, y0 + 1, y0, y0 + 1 };
  int xs[4] = { x0w, x0w, x1w, x1w };
  int id[4];
#pragma unroll
  for (int k = 0; k < 4; k++) {
    int yi = ys[k], xi = xs[k];
    bool pn = yi < 0, ps = yi >= H_FULL;
    if (pn) yi = -yi;
    if (ps) yi = 2 * H_FULL - yi;
    if (pn || ps) xi = (xi + W_FULL / 2) & (W_FULL - 1);
    yi = yi < 0 ? 0 : (yi > H_FULL - 1 ? H_FULL - 1 : yi);
    id[k] = yi * W_FULL + xi;
    atomicAdd(&den[id[k]], wt[k]);
  }
  idx4[m] = make_int4(id[0], id[1], id[2], id[3]);
  w4[m]   = make_float4(wt[0], wt[1], wt[2], wt[3]);
}

// ---------------- kernel 6: GEMM1 (576x128 @ 128x262144) with fused bilinear splat ----------------
// A = w_conv f16 (row-major, rows o = c_out*9 + tap). B = ht (column-major, K contiguous).
// Wave computes 16x16 tiles; A fragments held in registers for 256 columns.
__global__ void gemm1_splat(const _Float16* __restrict__ wA, const _Float16* __restrict__ ht,
                            const int4* __restrict__ idx4, const float4* __restrict__ w4,
                            float* __restrict__ num) {
  int w    = blockIdx.x * 8 + (threadIdx.x >> 5);  // global wave id, 36864 waves
  int lane = threadIdx.x & 31;
  int rowTile = w >> 10;                           // 0..35
  int chunk   = w & 1023;                          // 1024 chunks of 16 col-tiles
  int hi = lane >> 4, lm = lane & 15;

  // A fragments: lane holds row m = lane%16; half-select hi gives the documented
  // interleaved K pattern: [hi*8 .. hi*8+7] ++ [16+hi*8 .. 16+hi*8+7] per 32-K step.
  v16h afr[4];
  const _Float16* arow = wA + (size_t)(rowTile * 16 + lm) * CIN;
#pragma unroll
  for (int ks = 0; ks < 4; ks++) {
    const _Float16* p = arow + ks * 32 + hi * 8;
    afr[ks] = ld_frag2(p, p + 16);
  }

  for (int t = 0; t < 16; t++) {
    int ct  = chunk * 16 + t;
    int col = ct * 16 + lm;                        // B column = (b,hp,wp)
    const _Float16* bcol = ht + (size_t)col * CIN + hi * 8;
    __builtin_prefetch(ht + (size_t)(col + 16) * CIN, 0, 3);  // next col tile
    v8f acc = {};
#pragma unroll
    for (int ks = 0; ks < 4; ks++) {
      v16h bfr = ld_frag2(bcol + ks * 32, bcol + ks * 32 + 16);
      acc = __builtin_amdgcn_wmma_f32_16x16x32_f16(false, afr[ks], false, bfr,
                                                   (short)0, acc, false, false);
    }
    int b = col >> 17;                 // / HW
    int s = col & (HW - 1);
    // C/D layout: VGPR r holds row M = r + 8*(lane/16), column N = lane%16.
#pragma unroll
    for (int r = 0; r < 8; r++) {
      int o    = rowTile * 16 + r + 8 * hi;
      int cout = o / 9, tap = o - 9 * cout;
      int mg   = s * NTAP + tap;
      int4   id = idx4[mg];
      float4 wt = w4[mg];
      float  v  = acc[r];
      float* nb = num + ((size_t)(b * COUT + cout)) * HW2;
      atomicAdd(nb + id.x, v * wt.x);
      atomicAdd(nb + id.y, v * wt.y);
      atomicAdd(nb + id.z, v * wt.z);
      atomicAdd(nb + id.w, v * wt.w);
    }
  }
}

// ---------------- kernel 7: num/den + bias, transpose to K-contiguous f16 ----------------
// Output: zt[(b*HW2 + s)*64 + c]
__global__ void finalize_t(const float* __restrict__ num, const float* __restrict__ den,
                           const float* __restrict__ sbias, _Float16* __restrict__ zt) {
  __shared__ _Float16 tile[COUT * 64];
  int b  = blockIdx.x >> 13;                  // HW2/64 = 8192 tiles per batch
  int s0 = (blockIdx.x & 8191) * 64;
  for (int i = threadIdx.x; i < COUT * 64; i += 256) {
    int c = i >> 6, sl = i & 63;
    float d = den[s0 + sl]; d = d < 1e-8f ? 1e-8f : d;
    float v = num[((size_t)(b * COUT + c)) * HW2 + s0 + sl] / d + sbias[c];
    tile[i] = (_Float16)v;
  }
  __syncthreads();
  for (int i = threadIdx.x; i < COUT * 64; i += 256) {
    int sl = i >> 6, c = i & 63;
    zt[((size_t)(b * HW2 + s0 + sl)) * COUT + c] = tile[c * 64 + sl];
  }
}

// ---------------- kernel 8: GEMM2 (64x64 @ 64x1048576) + b_post ----------------
__global__ void gemm2(const _Float16* __restrict__ wP, const _Float16* __restrict__ zt,
                      const float* __restrict__ bpost, float* __restrict__ y) {
  int w    = blockIdx.x * 8 + (threadIdx.x >> 5);  // 16384 waves
  int lane = threadIdx.x & 31;
  int rowTile = w >> 12;                           // 0..3
  int chunk   = w & 4095;
  int hi = lane >> 4, lm = lane & 15;
  v16h afr[2];
  const _Float16* arow = wP + (size_t)(rowTile * 16 + lm) * COUT;
#pragma unroll
  for (int ks = 0; ks < 2; ks++) {
    const _Float16* p = arow + ks * 32 + hi * 8;
    afr[ks] = ld_frag2(p, p + 16);
  }
  for (int t = 0; t < 16; t++) {
    int ct  = chunk * 16 + t;
    int col = ct * 16 + lm;
    const _Float16* bcol = zt + (size_t)col * COUT + hi * 8;
    __builtin_prefetch(zt + (size_t)(col + 16) * COUT, 0, 3);
    v8f acc = {};
#pragma unroll
    for (int ks = 0; ks < 2; ks++) {
      v16h bfr = ld_frag2(bcol + ks * 32, bcol + ks * 32 + 16);
      acc = __builtin_amdgcn_wmma_f32_16x16x32_f16(false, afr[ks], false, bfr,
                                                   (short)0, acc, false, false);
    }
    int b = col >> 19;                 // / HW2
    int s = col & (HW2 - 1);
#pragma unroll
    for (int r = 0; r < 8; r++) {
      int o = rowTile * 16 + r + 8 * hi;
      y[((size_t)(b * COUT + o)) * HW2 + s] = acc[r] + bpost[o];
    }
  }
}

// ---------------- kernel 9: GN2 stats (per b, group of 2 channels) ----------------
__global__ void gn2_stats(const float* __restrict__ y, float* __restrict__ stats) {
  __shared__ float s1[256], s2[256];
  int blk = blockIdx.x;
  int b = blk >> 5, g = blk & 31;
  const float* base = y + ((size_t)(b * COUT + g * 2)) * HW2;   // 2*HW2 contiguous
  float a1 = 0.f, a2 = 0.f;
  for (int i = threadIdx.x; i < 2 * HW2; i += 256) { float v = base[i]; a1 += v; a2 += v * v; }
  s1[threadIdx.x] = a1; s2[threadIdx.x] = a2; __syncthreads();
  for (int off = 128; off > 0; off >>= 1) {
    if (threadIdx.x < off) { s1[threadIdx.x] += s1[threadIdx.x + off]; s2[threadIdx.x] += s2[threadIdx.x + off]; }
    __syncthreads();
  }
  if (threadIdx.x == 0) {
    float inv = 1.f / (2.f * HW2);
    float mu = s1[0] * inv;
    float var = s2[0] * inv - mu * mu;
    stats[blk * 2]     = mu;
    stats[blk * 2 + 1] = rsqrtf(var + 1e-5f);
  }
}

// ---------------- kernel 10: GN2 apply + exact GELU -> output ----------------
__global__ void gn2_apply_gelu(const float* __restrict__ y, const float* __restrict__ stats,
                               const float* __restrict__ gamma, const float* __restrict__ beta,
                               float* __restrict__ out) {
  long i  = ((long)blockIdx.x * 256 + threadIdx.x) * 4;
  int ch  = (int)(i >> 19);          // / HW2
  int o   = ch & 63;
  int b   = ch >> 6;
  float mu = stats[(b * GROUPS + (o >> 1)) * 2];
  float rs = stats[(b * GROUPS + (o >> 1)) * 2 + 1];
  float g = gamma[o], be = beta[o];
  float4 v = *(const float4*)(y + i);
  float4 r;
  r.x = gelu_exact((v.x - mu) * rs * g + be);
  r.y = gelu_exact((v.y - mu) * rs * g + be);
  r.z = gelu_exact((v.z - mu) * rs * g + be);
  r.w = gelu_exact((v.w - mu) * rs * g + be);
  *(float4*)(out + i) = r;
}

// ---------------- launch ----------------
extern "C" void kernel_launch(void* const* d_in, const int* in_sizes, int n_in,
                              void* d_out, int out_size, void* d_ws, size_t ws_size,
                              hipStream_t stream) {
  const float* x       = (const float*)d_in[0];
  const float* grid    = (const float*)d_in[1];
  const float* gamma1  = (const float*)d_in[2];
  const float* beta1   = (const float*)d_in[3];
  const float* w_conv  = (const float*)d_in[4];
  const float* sbias   = (const float*)d_in[5];
  const float* w_post  = (const float*)d_in[6];
  const float* b_post  = (const float*)d_in[7];
  const float* gamma2  = (const float*)d_in[8];
  const float* beta2   = (const float*)d_in[9];
  float* out = (float*)d_out;

  char* ws = (char*)d_ws;
  // 256B-aligned workspace layout (~510 MB total)
  _Float16* ht    = (_Float16*)(ws + 0);                         //  67,108,864 B
  _Float16* wc16  = (_Float16*)(ws + 67108864);                  //     147,456 B
  _Float16* wp16  = (_Float16*)(ws + 67256320);                  //       8,192 B
  int4*     idx4  = (int4*)    (ws + 67264512);                  //  18,874,368 B
  float4*   w4    = (float4*)  (ws + 86138880);                  //  18,874,368 B
  float*    num   = (float*)   (ws + 105013248);                 // 268,435,456 B (reused as y)
  float*    den   = (float*)   (ws + 373448704);                 //   2,097,152 B
  float*    st1   = (float*)   (ws + 375545856);                 //         512 B
  float*    st2   = (float*)   (ws + 375546368);                 //         512 B
  _Float16* zt    = (_Float16*)(ws + 375546880);                 // 134,217,728 B

  convert_w   <<<288,   256, 0, stream>>>(w_conv, w_post, wc16, wp16);
  gn1_stats   <<<64,    256, 0, stream>>>(x, st1);
  gn1_apply_t <<<8192,  256, 0, stream>>>(x, st1, gamma1, beta1, ht);
  // zero num (67,108,864 floats) and adjacent den (524,288 floats) in one pass
  zero_f32    <<<4096,  256, 0, stream>>>(num, 67633152L);
  splat_prep  <<<4608,  256, 0, stream>>>(grid, idx4, w4, den);
  gemm1_splat <<<4608,  256, 0, stream>>>(wc16, ht, idx4, w4, num);
  finalize_t  <<<16384, 256, 0, stream>>>(num, den, sbias, zt);
  gemm2       <<<2048,  256, 0, stream>>>(wp16, zt, b_post, num /* y aliases num */);
  gn2_stats   <<<64,    256, 0, stream>>>(num, st2);
  gn2_apply_gelu<<<65536,256, 0, stream>>>(num, st2, gamma2, beta2, out);
}
